// CFNetInteractionBlock_83373905150297
// MI455X (gfx1250) — matrix operand order repaired and checked by reading, compiled, not run on hardware
//
#include <hip/hip_runtime.h>

typedef __attribute__((ext_vector_type(16))) _Float16 v16h;
typedef __attribute__((ext_vector_type(8)))  float    v8f;

#define NFM 128
#define TILE_ROWS 64
#define THREADS 128

__device__ __forceinline__ float sspf(float x) {
  // shifted softplus: softplus(x) - log(2), numerically stable
  return fmaxf(x, 0.0f) + log1pf(__expf(-fabsf(x))) - 0.69314718055994530942f;
}

// ---------- fragment loaders ----------
// A fragment (16x32 f16) from row-major fp32 source, row stride NFM.
// 16-bit A 16x32 layout: lanes 0-15 -> M=lane, K in {0..7,16..23}; lanes 16-31 -> M=lane-16, K in {8..15,24..31}
__device__ __forceinline__ v16h load_a_f32g(const float* __restrict__ A, int row, int kc, int lane) {
  const int k1 = kc * 32 + ((lane & 16) ? 8 : 0);
  const float* p = A + (size_t)row * NFM + k1;
  float4 f0 = *reinterpret_cast<const float4*>(p);
  float4 f1 = *reinterpret_cast<const float4*>(p + 4);
  float4 f2 = *reinterpret_cast<const float4*>(p + 16);
  float4 f3 = *reinterpret_cast<const float4*>(p + 20);
  v16h r;
  r[0]=(_Float16)f0.x;  r[1]=(_Float16)f0.y;  r[2]=(_Float16)f0.z;  r[3]=(_Float16)f0.w;
  r[4]=(_Float16)f1.x;  r[5]=(_Float16)f1.y;  r[6]=(_Float16)f1.z;  r[7]=(_Float16)f1.w;
  r[8]=(_Float16)f2.x;  r[9]=(_Float16)f2.y;  r[10]=(_Float16)f2.z; r[11]=(_Float16)f2.w;
  r[12]=(_Float16)f3.x; r[13]=(_Float16)f3.y; r[14]=(_Float16)f3.z; r[15]=(_Float16)f3.w;
  return r;
}

// B fragment load from an LDS blob stored in exact fragment order: one contiguous 32B read per lane.
__device__ __forceinline__ v16h load_b_blob(const _Float16* sW, int nt, int kc, int lane) {
  return reinterpret_cast<const v16h*>(sW)[(nt * 4 + kc) * 32 + lane];
}
// A fragment load from an LDS blob in A-fragment order (per-wave region).
__device__ __forceinline__ v16h load_a_blob(const _Float16* sH, int wave, int kc, int lane) {
  return reinterpret_cast<const v16h*>(sH)[(wave * 4 + kc) * 32 + lane];
}

// 32 KB blob copy global->LDS
__device__ __forceinline__ void copy_blob(const _Float16* __restrict__ g, _Float16* s) {
  const uint4* src = reinterpret_cast<const uint4*>(g);
  uint4* dst = reinterpret_cast<uint4*>(s);
  for (int i = threadIdx.x; i < (NFM * NFM) / 8; i += blockDim.x) dst[i] = src[i];
}

// ---------- kernel 0: convert fp32 [K,N] weights -> f16 B-fragment-order blobs ----------
__global__ void k_prep(const float* w0, const float* w1, const float* w2, const float* w3, const float* w4,
                       _Float16* o0, _Float16* o1, _Float16* o2, _Float16* o3, _Float16* o4) {
  const float* src; _Float16* dst;
  switch (blockIdx.x) {
    case 0: src = w0; dst = o0; break;
    case 1: src = w1; dst = o1; break;
    case 2: src = w2; dst = o2; break;
    case 3: src = w3; dst = o3; break;
    default: src = w4; dst = o4; break;
  }
  for (int pos = threadIdx.x; pos < NFM * NFM; pos += blockDim.x) {
    int e  = pos & 15;
    int l  = (pos >> 4) & 31;
    int f  = pos >> 9;            // nt*4 + kc
    int kc = f & 3, nt = f >> 2;
    int n  = nt * 16 + (l & 15);
    int k  = kc * 32 + ((l & 16) ? 8 : 0) + (e & 7) + ((e & 8) ? 16 : 0);
    dst[pos] = (_Float16)src[k * NFM + n];   // W is [in(K), out(N)] row-major
  }
}

__global__ void k_zero(float* p, long long n) {
  long long i = (long long)blockIdx.x * blockDim.x + threadIdx.x;
  if (i < n) p[i] = 0.0f;
}

// ---------- kernel 1: F = X @ W_in2fac ----------
__global__ __launch_bounds__(THREADS) void k_gemm_f(const float* __restrict__ X,
                                                    const _Float16* __restrict__ Wblob,
                                                    float* __restrict__ F, int nrows) {
  __shared__ alignas(32) _Float16 sW[NFM * NFM];
  copy_blob(Wblob, sW);
  __syncthreads();
  const int lane = threadIdx.x & 31, wave = threadIdx.x >> 5;
  const int strip = blockIdx.x * TILE_ROWS + wave * 16;
  const bool full = (strip + 16 <= nrows);   // wave-uniform -> scalar branch
  int row = strip + (lane & 15); if (row >= nrows) row = nrows - 1;
  v16h a[4];
#pragma unroll
  for (int kc = 0; kc < 4; ++kc) a[kc] = load_a_f32g(X, row, kc, lane);
#pragma unroll
  for (int nt = 0; nt < 8; ++nt) {
    v8f acc = {};
#pragma unroll
    for (int kc = 0; kc < 4; ++kc) {
      v16h b = load_b_blob(sW, nt, kc, lane);
      acc = __builtin_amdgcn_wmma_f32_16x16x32_f16(false, a[kc], false, b, (short)0, acc, false, false);
    }
    const int cc = nt * 16 + (lane & 15);
    float* dst = F + (size_t)(strip + ((lane & 16) ? 8 : 0)) * NFM + cc;
    if (full) {
#pragma unroll
      for (int r = 0; r < 8; ++r) dst[(size_t)r * NFM] = acc[r];
    } else {
      const int rb = strip + ((lane & 16) ? 8 : 0);
#pragma unroll
      for (int r = 0; r < 8; ++r) if (rb + r < nrows) dst[(size_t)r * NFM] = acc[r];
    }
  }
}

// ---------- kernel 2: fused filter-net + cfconv scatter ----------
// w = ssp(ssp(dijk@W1+b1)@W2+b2); conv[seg_i[e]] += w[t]*F[idx_j[e]], e=seg_j[t]
__global__ __launch_bounds__(THREADS) void k_filter_conv(
    const float* __restrict__ Dijk,
    const _Float16* __restrict__ W1blob, const _Float16* __restrict__ W2blob,
    const float* __restrict__ b1, const float* __restrict__ b2,
    const int* __restrict__ seg_j, const int* __restrict__ idx_j, const int* __restrict__ seg_i,
    const float* __restrict__ F, float* __restrict__ Conv, int nT) {
  __shared__ alignas(32) float sR0[TILE_ROWS * NFM];   // 32KB: W1 blob -> h frags -> w fp32
  __shared__ alignas(32) _Float16 sW2[NFM * NFM];      // 32KB: W2 blob -> (i,j) index stage
  _Float16* sR0h = reinterpret_cast<_Float16*>(sR0);
  copy_blob(W1blob, sR0h);
  copy_blob(W2blob, sW2);
  __syncthreads();

  const int lane = threadIdx.x & 31, wave = threadIdx.x >> 5;
  const int tile = blockIdx.x * TILE_ROWS;
  const int strip = tile + wave * 16;
  int rowg = strip + (lane & 15); if (rowg >= nT) rowg = nT - 1;

  v16h a[4];
#pragma unroll
  for (int kc = 0; kc < 4; ++kc) a[kc] = load_a_f32g(Dijk, rowg, kc, lane);

  // GEMM1: h = dijk @ W1 + b1
  v8f hacc[8];
#pragma unroll
  for (int nt = 0; nt < 8; ++nt) {
    float bv = b1[nt * 16 + (lane & 15)];
    v8f acc;
#pragma unroll
    for (int r = 0; r < 8; ++r) acc[r] = bv;
#pragma unroll
    for (int kc = 0; kc < 4; ++kc) {
      v16h b = load_b_blob(sR0h, nt, kc, lane);
      acc = __builtin_amdgcn_wmma_f32_16x16x32_f16(false, a[kc], false, b, (short)0, acc, false, false);
    }
    hacc[nt] = acc;
  }
  __syncthreads();  // everyone done reading W1 blob

  // ssp(h) -> A-fragment-order f16 blob (per-wave region of sR0h); base hoisted, stride 32B
#pragma unroll
  for (int nt = 0; nt < 8; ++nt) {
    const int col = nt * 16 + (lane & 15);
    const int kc = col >> 5, kr = col & 31;
    const int dl0 = ((lane & 16) ? 8 : 0) + (((kr >> 3) & 1) << 4);
    const int e   = (kr & 7) + (((kr >> 4) & 1) << 3);
    _Float16* base = sR0h + (((wave * 4 + kc) * 32 + dl0) << 4) + e;
#pragma unroll
    for (int r = 0; r < 8; ++r) base[r << 4] = (_Float16)sspf(hacc[nt][r]);
  }

  // GEMM2: w = h @ W2 + b2  (A from own-wave LDS region)
#pragma unroll
  for (int kc = 0; kc < 4; ++kc) a[kc] = load_a_blob(sR0h, wave, kc, lane);
  v8f wacc[8];
#pragma unroll
  for (int nt = 0; nt < 8; ++nt) {
    float bv = b2[nt * 16 + (lane & 15)];
    v8f acc;
#pragma unroll
    for (int r = 0; r < 8; ++r) acc[r] = bv;
#pragma unroll
    for (int kc = 0; kc < 4; ++kc) {
      v16h b = load_b_blob(sW2, nt, kc, lane);
      acc = __builtin_amdgcn_wmma_f32_16x16x32_f16(false, a[kc], false, b, (short)0, acc, false, false);
    }
    wacc[nt] = acc;
  }
  __syncthreads();  // h reads + W2 reads complete; both LDS regions reusable now

  // stage per-row (i, j) indices in the dead W2 region; prefetch the F rows we'll gather
  int* sIdx = reinterpret_cast<int*>(sW2);
  if (threadIdx.x < TILE_ROWS) {
    int t = tile + threadIdx.x;
    int i = -1, j = 0;
    if (t < nT) { int e = seg_j[t]; j = idx_j[e]; i = seg_i[e]; }
    sIdx[threadIdx.x] = i;
    sIdx[TILE_ROWS + threadIdx.x] = j;
    if (i >= 0) __builtin_prefetch(&F[(size_t)j * NFM], 0, 0);  // global_prefetch_b8
  }

  // ssp(w) -> fp32 row-major tile for the scatter phase
#pragma unroll
  for (int nt = 0; nt < 8; ++nt) {
    const int col = nt * 16 + (lane & 15);
    const int rb = wave * 16 + ((lane & 16) ? 8 : 0);
#pragma unroll
    for (int r = 0; r < 8; ++r) sR0[(rb + r) * NFM + col] = sspf(wacc[nt][r]);
  }
  __syncthreads();

  // fused gather-multiply-scatter; run-length combine on sorted seg_i to cut atomics ~16x.
  // i/j go through readfirstlane -> SGPRs: scalar flush branch + saddr-form addressing.
  const int c = threadIdx.x;  // feature column 0..127
  float run = 0.0f; int runI = -1;
  for (int r = 0; r < TILE_ROWS; ++r) {
    int i = __builtin_amdgcn_readfirstlane(sIdx[r]);
    if (i < 0) break;  // tail rows
    int j = __builtin_amdgcn_readfirstlane(sIdx[TILE_ROWS + r]);
    float val = sR0[r * NFM + c] * F[(size_t)j * NFM + c];
    if (i == runI) { run += val; }
    else {
      if (runI >= 0) atomicAdd(&Conv[(size_t)runI * NFM + c], run);
      runI = i; run = val;
    }
  }
  if (runI >= 0) atomicAdd(&Conv[(size_t)runI * NFM + c], run);
}

// ---------- kernel 3: c = ssp(conv@Wfac+bfac); v = c@Wdense+bdense; y = x + v ----------
__global__ __launch_bounds__(THREADS) void k_out(
    const float* __restrict__ Conv, const float* __restrict__ X,
    const _Float16* __restrict__ Wfacblob, const _Float16* __restrict__ Wdenseblob,
    const float* __restrict__ bfac, const float* __restrict__ bdense,
    float* __restrict__ Y, float* __restrict__ V, int nrows) {
  __shared__ alignas(32) float sR0[TILE_ROWS * NFM];   // Wfac blob -> c frags
  __shared__ alignas(32) _Float16 sW2[NFM * NFM];
  _Float16* sR0h = reinterpret_cast<_Float16*>(sR0);
  copy_blob(Wfacblob, sR0h);
  copy_blob(Wdenseblob, sW2);
  __syncthreads();

  const int lane = threadIdx.x & 31, wave = threadIdx.x >> 5;
  const int strip = blockIdx.x * TILE_ROWS + wave * 16;
  const bool full = (strip + 16 <= nrows);   // wave-uniform
  int row = strip + (lane & 15); if (row >= nrows) row = nrows - 1;

  v16h a[4];
#pragma unroll
  for (int kc = 0; kc < 4; ++kc) a[kc] = load_a_f32g(Conv, row, kc, lane);

  v8f cacc[8];
#pragma unroll
  for (int nt = 0; nt < 8; ++nt) {
    float bv = bfac[nt * 16 + (lane & 15)];
    v8f acc;
#pragma unroll
    for (int r = 0; r < 8; ++r) acc[r] = bv;
#pragma unroll
    for (int kc = 0; kc < 4; ++kc) {
      v16h b = load_b_blob(sR0h, nt, kc, lane);
      acc = __builtin_amdgcn_wmma_f32_16x16x32_f16(false, a[kc], false, b, (short)0, acc, false, false);
    }
    cacc[nt] = acc;
  }
  __syncthreads();

#pragma unroll
  for (int nt = 0; nt < 8; ++nt) {
    const int col = nt * 16 + (lane & 15);
    const int kc = col >> 5, kr = col & 31;
    const int dl0 = ((lane & 16) ? 8 : 0) + (((kr >> 3) & 1) << 4);
    const int e   = (kr & 7) + (((kr >> 4) & 1) << 3);
    _Float16* base = sR0h + (((wave * 4 + kc) * 32 + dl0) << 4) + e;
#pragma unroll
    for (int r = 0; r < 8; ++r) base[r << 4] = (_Float16)sspf(cacc[nt][r]);
  }

#pragma unroll
  for (int kc = 0; kc < 4; ++kc) a[kc] = load_a_blob(sR0h, wave, kc, lane);

#pragma unroll
  for (int nt = 0; nt < 8; ++nt) {
    float bv = bdense[nt * 16 + (lane & 15)];
    v8f acc;
#pragma unroll
    for (int r = 0; r < 8; ++r) acc[r] = bv;
#pragma unroll
    for (int kc = 0; kc < 4; ++kc) {
      v16h b = load_b_blob(sW2, nt, kc, lane);
      acc = __builtin_amdgcn_wmma_f32_16x16x32_f16(false, a[kc], false, b, (short)0, acc, false, false);
    }
    const int cc = nt * 16 + (lane & 15);
    const size_t rowoff = (size_t)(strip + ((lane & 16) ? 8 : 0)) * NFM + cc;
    if (full) {
#pragma unroll
      for (int r = 0; r < 8; ++r) {
        float v = acc[r];
        V[rowoff + (size_t)r * NFM] = v;
        Y[rowoff + (size_t)r * NFM] = X[rowoff + (size_t)r * NFM] + v;
      }
    } else {
      const int rb = strip + ((lane & 16) ? 8 : 0);
#pragma unroll
      for (int r = 0; r < 8; ++r) {
        if (rb + r < nrows) {
          float v = acc[r];
          V[rowoff + (size_t)r * NFM] = v;
          Y[rowoff + (size_t)r * NFM] = X[rowoff + (size_t)r * NFM] + v;
        }
      }
    }
  }
}

extern "C" void kernel_launch(void* const* d_in, const int* in_sizes, int n_in,
                              void* d_out, int out_size, void* d_ws, size_t ws_size,
                              hipStream_t stream) {
  const float* x        = (const float*)d_in[0];
  const float* dijk     = (const float*)d_in[1];
  const int*   idx_j    = (const int*)d_in[2];
  const int*   seg_i    = (const int*)d_in[3];
  const int*   seg_j    = (const int*)d_in[4];
  // d_in[5] seg_i_sum unused by the reference math
  const float* W_f1     = (const float*)d_in[6];
  const float* b_f1     = (const float*)d_in[7];
  const float* W_f2     = (const float*)d_in[8];
  const float* b_f2     = (const float*)d_in[9];
  const float* W_in2fac = (const float*)d_in[10];
  const float* W_fac2out= (const float*)d_in[11];
  const float* b_fac2out= (const float*)d_in[12];
  const float* W_dense  = (const float*)d_in[13];
  const float* b_dense  = (const float*)d_in[14];

  const int nAtoms = in_sizes[0] / NFM;
  const int nT     = in_sizes[2];

  float* Y = (float*)d_out;
  float* V = Y + (size_t)nAtoms * NFM;

  char* ws = (char*)d_ws;
  _Float16* Wt_f1     = (_Float16*)(ws + 0);
  _Float16* Wt_f2     = (_Float16*)(ws + 32768);
  _Float16* Wt_in2fac = (_Float16*)(ws + 65536);
  _Float16* Wt_fac    = (_Float16*)(ws + 98304);
  _Float16* Wt_dense  = (_Float16*)(ws + 131072);
  float*    F         = (float*)(ws + 163840);
  float*    Conv      = F + (size_t)nAtoms * NFM;

  k_prep<<<5, 256, 0, stream>>>(W_f1, W_f2, W_in2fac, W_fac2out, W_dense,
                                Wt_f1, Wt_f2, Wt_in2fac, Wt_fac, Wt_dense);
  long long convN = (long long)nAtoms * NFM;
  k_zero<<<(unsigned)((convN + 255) / 256), 256, 0, stream>>>(Conv, convN);
  k_gemm_f<<<(nAtoms + TILE_ROWS - 1) / TILE_ROWS, THREADS, 0, stream>>>(x, Wt_in2fac, F, nAtoms);
  k_filter_conv<<<(nT + TILE_ROWS - 1) / TILE_ROWS, THREADS, 0, stream>>>(
      dijk, Wt_f1, Wt_f2, b_f1, b_f2, seg_j, idx_j, seg_i, F, Conv, nT);
  k_out<<<(nAtoms + TILE_ROWS - 1) / TILE_ROWS, THREADS, 0, stream>>>(
      Conv, x, Wt_fac, Wt_dense, b_fac2out, b_dense, Y, V, nAtoms);
}
